// NPF_1829656068460
// MI455X (gfx1250) — matrix operand order repaired and checked by reading, not compiled, not run on hardware
//
#include <hip/hip_runtime.h>

// Problem constants (from reference): B=1024, I=512, O=64, N=2048, C=256, S=8
#define B_SZ 1024
#define I_SZ 512
#define O_SZ 64
#define N_SZ 2048
#define C_SZ 256
#define S_SZ 8
#define BT   128              // batch columns per block
#define WS   (I_SZ + 8)       // padded LDS row stride (bf16 elems): rows 4 banks apart -> conflict-free b128

typedef __bf16        bf16x16 __attribute__((ext_vector_type(16)));
typedef float         f32x8   __attribute__((ext_vector_type(8)));
typedef float         f32x4   __attribute__((ext_vector_type(4)));
typedef unsigned int  u32x4   __attribute__((ext_vector_type(4)));

union Frag { u32x4 u[2]; bf16x16 v; };

__device__ __forceinline__ unsigned short f2bf(float f) {
  union { float f; unsigned u; } x; x.f = f;
  unsigned u = x.u;
  return (unsigned short)((u + 0x7FFFu + ((u >> 16) & 1u)) >> 16);  // RNE truncate
}
__device__ __forceinline__ unsigned pack2(float a, float b) {
  return (unsigned)f2bf(a) | ((unsigned)f2bf(b) << 16);
}

__device__ __forceinline__ void load_frag(Frag& f, const unsigned short* row,
                                          int k0, int hf8) {
  f.u[0] = *reinterpret_cast<const u32x4*>(row + k0 + hf8);
  f.u[1] = *reinterpret_cast<const u32x4*>(row + k0 + 16 + hf8);
}
__device__ __forceinline__ void load_fragB(Frag& f, const unsigned short* row,
                                           int k0, int hf16) {
  f.u[0] = *reinterpret_cast<const u32x4*>(row + k0 + hf16 / 2);
  f.u[1] = *reinterpret_cast<const u32x4*>(row + k0 + 16 + hf16 / 2);
}

__global__ __launch_bounds__(256)
void npf_committee_kernel(const float* __restrict__ x,
                          const float* __restrict__ neurons,
                          const int*   __restrict__ npf,
                          float* __restrict__ out) {
  extern __shared__ unsigned short smem[];
  unsigned short* wsh0 = smem;                       // O_SZ x WS (bf16 weights, buffer 0)
  unsigned short* wsh1 = smem + O_SZ * WS;           // O_SZ x WS (bf16 weights, buffer 1)
  unsigned short* xsh  = smem + 2 * O_SZ * WS;       // BT x WS  (bf16 x-tile)

  const int c    = blockIdx.y;
  const int b0   = blockIdx.x * BT;
  const int tid  = threadIdx.x;
  const int wave = tid >> 5;
  const int lane = tid & 31;
  const int hf   = lane >> 4;          // K-half selector per WMMA operand layout
  const int r    = lane & 15;          // row (A) / column (B) within 16-wide tile
  const int hf8  = hf * 8;
  const int hf16 = hf * 16;
  // 2x2 macro-tile per wave: 32 o x 32 b
  const int otile = (wave >> 2) * 32;  // 0 or 32
  const int btile = (wave & 3) * 32;   // 0,32,64,96

  // ---- stage x tile (BT x I) fp32 -> bf16 into LDS (once per block) ----
  for (int idx = tid; idx < BT * (I_SZ / 4); idx += 256) {
    const int row  = idx >> 7;         // / (I/4 = 128)
    const int col4 = idx & 127;
    const f32x4 v = *reinterpret_cast<const f32x4*>(
        x + (size_t)(b0 + row) * I_SZ + (size_t)col4 * 4);
    unsigned* dst = reinterpret_cast<unsigned*>(xsh + row * WS + col4 * 4);
    dst[0] = pack2(v[0], v[1]);
    dst[1] = pack2(v[2], v[3]);
  }

  // ---- stage w[npf[c,0]] into buffer 0 ----
  {
    const size_t nb = (size_t)npf[c * S_SZ] * (size_t)(O_SZ * I_SZ);
    for (int idx = tid; idx < O_SZ * (I_SZ / 4); idx += 256) {
      const int row  = idx >> 7;
      const int col4 = idx & 127;
      const f32x4 v = *reinterpret_cast<const f32x4*>(
          neurons + nb + (size_t)row * I_SZ + (size_t)col4 * 4);
      unsigned* dst = reinterpret_cast<unsigned*>(wsh0 + row * WS + col4 * 4);
      dst[0] = pack2(v[0], v[1]);
      dst[1] = pack2(v[2], v[3]);
    }
  }
  __syncthreads();

  float prod[4][8];
#pragma unroll
  for (int t = 0; t < 4; ++t)
#pragma unroll
    for (int i = 0; i < 8; ++i) prod[t][i] = 1.0f;

  const unsigned short* brow0 = xsh + (btile + r) * WS;
  const unsigned short* brow1 = brow0 + 16 * WS;

  for (int s = 0; s < S_SZ; ++s) {
    // warm L2 for s+2's weights (global_prefetch_b8)
    if (s + 2 < S_SZ) {
      const float* pf = neurons +
          (size_t)npf[c * S_SZ + s + 2] * (size_t)(O_SZ * I_SZ);
#pragma unroll
      for (int j = 0; j < 4; ++j)   // 1024 cachelines of 128B, 256 threads x 4
        __builtin_prefetch(pf + (size_t)(tid + j * 256) * 32, 0, 0);
    }

    // ---- stage w[npf[c,s+1]] into alternate buffer (overlaps with compute) ----
    if (s + 1 < S_SZ) {
      unsigned short* wnext = ((s + 1) & 1) ? wsh1 : wsh0;
      const size_t nb = (size_t)npf[c * S_SZ + s + 1] * (size_t)(O_SZ * I_SZ);
      for (int idx = tid; idx < O_SZ * (I_SZ / 4); idx += 256) {
        const int row  = idx >> 7;
        const int col4 = idx & 127;
        const f32x4 v = *reinterpret_cast<const f32x4*>(
            neurons + nb + (size_t)row * I_SZ + (size_t)col4 * 4);
        unsigned* dst = reinterpret_cast<unsigned*>(wnext + row * WS + col4 * 4);
        dst[0] = pack2(v[0], v[1]);
        dst[1] = pack2(v[2], v[3]);
      }
    }

    // ---- compute s from current buffer: 2x2 tiles, 4 WMMAs per K-step ----
    const unsigned short* wb = (s & 1) ? wsh1 : wsh0;
    const unsigned short* arow0 = wb + (otile + r) * WS;
    const unsigned short* arow1 = arow0 + 16 * WS;

    f32x8 acc00 = {0.f,0.f,0.f,0.f,0.f,0.f,0.f,0.f};
    f32x8 acc01 = acc00, acc10 = acc00, acc11 = acc00;

#pragma unroll
    for (int k0 = 0; k0 < I_SZ; k0 += 32) {
      Frag a0, a1, bb0, bb1;
      load_frag (a0,  arow0, k0, hf8);
      load_frag (a1,  arow1, k0, hf8);
      load_fragB(bb0, brow0, k0, hf16);
      load_fragB(bb1, brow1, k0, hf16);
      acc00 = __builtin_amdgcn_wmma_f32_16x16x32_bf16(false, a0.v, false, bb0.v, (short)0, acc00, false, false);
      acc01 = __builtin_amdgcn_wmma_f32_16x16x32_bf16(false, a0.v, false, bb1.v, (short)0, acc01, false, false);
      acc10 = __builtin_amdgcn_wmma_f32_16x16x32_bf16(false, a1.v, false, bb0.v, (short)0, acc10, false, false);
      acc11 = __builtin_amdgcn_wmma_f32_16x16x32_bf16(false, a1.v, false, bb1.v, (short)0, acc11, false, false);
    }

    // sigmoid(z) = 1 / (1 + 2^(-z*log2 e)) ; v_exp_f32 + v_rcp_f32
#pragma unroll
    for (int v = 0; v < 8; ++v) {
      prod[0][v] *= __builtin_amdgcn_rcpf(1.0f + __builtin_amdgcn_exp2f(-acc00[v] * 1.44269504f));
      prod[1][v] *= __builtin_amdgcn_rcpf(1.0f + __builtin_amdgcn_exp2f(-acc01[v] * 1.44269504f));
      prod[2][v] *= __builtin_amdgcn_rcpf(1.0f + __builtin_amdgcn_exp2f(-acc10[v] * 1.44269504f));
      prod[3][v] *= __builtin_amdgcn_rcpf(1.0f + __builtin_amdgcn_exp2f(-acc11[v] * 1.44269504f));
    }

    __syncthreads();  // seals: compute(s) done reading wb; staging(s+1) visible for next iter
  }

  // D layout: lane owns b = b0+btile+bb*16+r, o = otile+oo*16+hf*8+{0..7} (contiguous in [B,C,O])
#pragma unroll
  for (int oo = 0; oo < 2; ++oo) {
#pragma unroll
    for (int bb = 0; bb < 2; ++bb) {
      const float* p = prod[oo * 2 + bb];
      float* outp = out +
          ((size_t)(b0 + btile + bb * 16 + r) * C_SZ + c) * O_SZ +
          otile + oo * 16 + hf8;
      f32x4 lo = { p[0], p[1], p[2], p[3] };
      f32x4 hi = { p[4], p[5], p[6], p[7] };
      *reinterpret_cast<f32x4*>(outp)     = lo;
      *reinterpret_cast<f32x4*>(outp + 4) = hi;
    }
  }
}

extern "C" void kernel_launch(void* const* d_in, const int* in_sizes, int n_in,
                              void* d_out, int out_size, void* d_ws, size_t ws_size,
                              hipStream_t stream) {
  (void)in_sizes; (void)n_in; (void)out_size; (void)d_ws; (void)ws_size;
  const float* x       = (const float*)d_in[0];
  const float* neurons = (const float*)d_in[1];
  const int*   npf     = (const int*)d_in[2];
  float*       out     = (float*)d_out;

  dim3 grid(B_SZ / BT, C_SZ);     // (8, 256) = 2048 blocks
  dim3 block(256);                // 8 wave32s; each wave owns a 32x32 macro-tile
  size_t shmem = (size_t)(2 * O_SZ + BT) * WS * sizeof(unsigned short);  // 266,240 B < 320 KB
  npf_committee_kernel<<<grid, block, shmem, stream>>>(x, neurons, npf, out);
}
